// VirtualNode_67345087201310
// MI455X (gfx1250) — compile-verified, matrix operand order checked by reading, not run on hardware
//
#include <hip/hip_runtime.h>
#include <hip/hip_bf16.h>
#include <stdint.h>

// ---------------------------------------------------------------------------
// MI455X (gfx1250) VirtualNode layer.
// Dominant cost is [N,512]@[512,256] attention GEMM (131 GFLOP) -> done with
// v_wmma_f32_16x16x32_bf16 (bf16 A/B, f32 accum). Epilogue tanh uses the
// gfx1250 hardware V_TANH_F32 transcendental. Everything else is
// bandwidth-bound and fused around it.
// ---------------------------------------------------------------------------

typedef __attribute__((ext_vector_type(16))) __bf16 v16bf;
typedef __attribute__((ext_vector_type(8)))  float  v8f;

#define NN   500000
#define HH   256
#define HH2  512
#define GG   1024
#define EPSF 1e-5f
#define LDK  264       // padded LDS K-stride (bf16): 528B rows -> 16B bank skew

// ---- WMMA fragment helpers (layout per CDNA5 ISA, 16-bit A 16x32:
// lane<16 holds K {k0..k0+7, k0+16..k0+23}; lane>=16 holds {k0+8..15, k0+24..31})
__device__ __forceinline__ v16bf frag_lds(const __bf16* As, int rowBase, int k0) {
    int lane = threadIdx.x & 31;
    const __bf16* p = As + (rowBase + (lane & 15)) * LDK + k0 + ((lane >> 4) << 3);
    union { uint4 u[2]; v16bf v; } t;
    t.u[0] = *(const uint4*)(p);
    t.u[1] = *(const uint4*)(p + 16);
    return t.v;
}
__device__ __forceinline__ v16bf frag_gbl(const __bf16* BT, int colBase, int k0, int K) {
    int lane = threadIdx.x & 31;
    const __bf16* p = BT + (size_t)(colBase + (lane & 15)) * K + k0 + ((lane >> 4) << 3);
    union { uint4 u[2]; v16bf v; } t;
    t.u[0] = *(const uint4*)(p);
    t.u[1] = *(const uint4*)(p + 16);
    return t.v;
}
__device__ __forceinline__ v8f wmma_bf16(v16bf a, v16bf b, v8f c) {
    return __builtin_amdgcn_wmma_f32_16x16x32_bf16(false, a, false, b, (short)0, c, false, false);
}

// gfx1250 hardware tanh (TRANS op). The trailing v_nop satisfies the ISA's
// "1 independent op after a TRANS op before its output is consumed" rule,
// which the compiler cannot enforce across an asm boundary.
__device__ __forceinline__ float hw_tanh(float x) {
    float r;
    asm volatile("v_tanh_f32 %0, %1\n\tv_nop" : "=v"(r) : "v"(x));
    return r;
}

// ---------------------------------------------------------------------------
__global__ void k_zero(float* p, int n) {
    int i = blockIdx.x * blockDim.x + threadIdx.x;
    if (i < n) p[i] = 0.f;
}

// Segment sum over sorted batch: 256 threads = 256 cols, 128 rows per block,
// register accumulation, atomic flush only at segment boundaries.
__global__ void k_segsum(const float* __restrict__ x, const int* __restrict__ batch,
                         float* __restrict__ sums, float* __restrict__ counts) {
    int t  = threadIdx.x;
    int r0 = blockIdx.x * 128;
    int cur = -1; float acc = 0.f, cnt = 0.f;
    for (int r = 0; r < 128; ++r) {
        int row = r0 + r;
        if (row >= NN) break;
        int b = batch[row];
        if (b != cur) {
            if (cur >= 0) {
                atomicAdd(&sums[cur * HH + t], acc);
                if (t == 0) atomicAdd(&counts[cur], cnt);
            }
            cur = b; acc = 0.f; cnt = 0.f;
        }
        acc += x[(size_t)row * HH + t];
        cnt += 1.f;
    }
    if (cur >= 0) {
        atomicAdd(&sums[cur * HH + t], acc);
        if (t == 0) atomicAdd(&counts[cur], cnt);
    }
}

// dst[o*K + k] = bf16(src[k*O + o])  (pre-transpose weights to [out][K] bf16)
__global__ void k_transpose_bf16(const float* __restrict__ src, __bf16* __restrict__ dst,
                                 int K, int O) {
    int o = blockIdx.x;
    for (int k = threadIdx.x; k < K; k += blockDim.x)
        dst[(size_t)o * K + k] = (__bf16)src[(size_t)k * O + o];
}

// a[g][h] = virtual_emb[h] + sums[g][h] / max(counts[g], 1)
__global__ void k_vnpre(const float* __restrict__ sums, const float* __restrict__ counts,
                        const float* __restrict__ ve, float* __restrict__ a) {
    int g = blockIdx.x, h = threadIdx.x;
    float c = counts[g]; c = c > 1.f ? c : 1.f;
    a[g * HH + h] = ve[h] + sums[g * HH + h] / c;
}

// GEMM1: h1[1024,512] = a[1024,256] @ W1 + b1.  grid(16,2), 128 thr (4 waves).
__global__ void k_gemm1(const float* __restrict__ A, const __bf16* __restrict__ BT,
                        const float* __restrict__ bias, float* __restrict__ C) {
    __shared__ __bf16 As[64 * LDK];
    int m0 = blockIdx.x * 64;
    int cb = blockIdx.y * 256;
    for (int i = threadIdx.x; i < 64 * 256; i += 128) {
        int r = i >> 8, c = i & 255;
        As[r * LDK + c] = (__bf16)A[(size_t)(m0 + r) * HH + c];
    }
    __syncthreads();
    int wave = threadIdx.x >> 5, lane = threadIdx.x & 31;
    int rowBase = wave * 16;
    for (int ct = 0; ct < 16; ++ct) {
        int c0 = cb + ct * 16;
        v8f acc = {};
        #pragma unroll
        for (int k0 = 0; k0 < 256; k0 += 32)
            acc = wmma_bf16(frag_lds(As, rowBase, k0), frag_gbl(BT, c0, k0, HH), acc);
        int col = c0 + (lane & 15);
        float bv = bias[col];
        int rbase = m0 + rowBase + ((lane >> 4) << 3);
        #pragma unroll
        for (int v = 0; v < 8; ++v)
            C[(size_t)(rbase + v) * HH2 + col] = acc[v] + bv;
    }
}

// BatchNorm training stats per column of h1 -> scale/shift (biased var).
__global__ void k_bnstats(const float* __restrict__ h1, const float* __restrict__ gamma,
                          const float* __restrict__ beta, float* __restrict__ scale,
                          float* __restrict__ shift) {
    int c = blockIdx.x * blockDim.x + threadIdx.x;
    if (c >= HH2) return;
    float s = 0.f, sq = 0.f;
    for (int r = 0; r < GG; ++r) { float v = h1[(size_t)r * HH2 + c]; s += v; sq += v * v; }
    float mu  = s  * (1.f / GG);
    float var = sq * (1.f / GG) - mu * mu;
    float sc  = gamma[c] * rsqrtf(var + EPSF);
    scale[c] = sc;
    shift[c] = beta[c] - mu * sc;
}

// GEMM2: vn = relu(BN(h1)) @ W2 + b2.  BN+ReLU fused into the LDS A-load.
// Writes f32 vn_updated (output #2) and a bf16 copy for the node-gather.
__global__ void k_gemm2(const float* __restrict__ h1, const float* __restrict__ scale,
                        const float* __restrict__ shift, const __bf16* __restrict__ BT,
                        const float* __restrict__ bias, float* __restrict__ vn_out,
                        __bf16* __restrict__ vnb) {
    __shared__ __bf16 As[64 * LDK];
    int m0 = blockIdx.x * 64;
    int wave = threadIdx.x >> 5, lane = threadIdx.x & 31;
    int rowBase = wave * 16;
    v8f acc[16];
    #pragma unroll
    for (int i = 0; i < 16; ++i) acc[i] = (v8f){};
    for (int kc = 0; kc < 2; ++kc) {
        if (kc) __syncthreads();
        for (int i = threadIdx.x; i < 64 * 256; i += 128) {
            int r = i >> 8, c = i & 255;
            int kcol = kc * 256 + c;
            float v = h1[(size_t)(m0 + r) * HH2 + kcol] * scale[kcol] + shift[kcol];
            As[r * LDK + c] = (__bf16)(v > 0.f ? v : 0.f);
        }
        __syncthreads();
        #pragma unroll
        for (int ct = 0; ct < 16; ++ct)
            #pragma unroll
            for (int k0 = 0; k0 < 256; k0 += 32)
                acc[ct] = wmma_bf16(frag_lds(As, rowBase, k0),
                                    frag_gbl(BT, ct * 16, kc * 256 + k0, HH2), acc[ct]);
    }
    int rbase = m0 + rowBase + ((lane >> 4) << 3);
    #pragma unroll
    for (int ct = 0; ct < 16; ++ct) {
        int col = ct * 16 + (lane & 15);
        float bv = bias[col];
        #pragma unroll
        for (int v = 0; v < 8; ++v) {
            float val = acc[ct][v] + bv;
            vn_out[(size_t)(rbase + v) * HH + col] = val;
            vnb[(size_t)(rbase + v) * HH + col]    = (__bf16)val;
        }
    }
}

// Attention scores: s[i] = tanh([x[i] ; vn[batch[i]]] @ Wa1 + ba1) . Wa2
// (ba2 omitted: constant shift is softmax-invariant). 64 rows/block, 4 waves,
// 16x16 col tiles, K=512 staged in two 256-wide LDS chunks.
__global__ void k_scores(const float* __restrict__ x, const int* __restrict__ batch,
                         const __bf16* __restrict__ vnb, const __bf16* __restrict__ Wa1T,
                         const float* __restrict__ ba1, const float* __restrict__ wa2,
                         float* __restrict__ scores) {
    __shared__ __bf16 As[64 * LDK];
    __shared__ int bsh[64];
    int m0 = blockIdx.x * 64;
    if (threadIdx.x < 64) {
        int row = m0 + threadIdx.x;
        bsh[threadIdx.x] = (row < NN) ? batch[row] : 0;
    }
    __syncthreads();
    int wave = threadIdx.x >> 5, lane = threadIdx.x & 31;
    int rowBase = wave * 16;
    v8f acc[16];
    #pragma unroll
    for (int i = 0; i < 16; ++i) acc[i] = (v8f){};
    for (int kc = 0; kc < 2; ++kc) {
        if (kc) __syncthreads();
        if (kc == 0) {
            for (int i = threadIdx.x; i < 64 * 256; i += 128) {
                int r = i >> 8, c = i & 255;
                int row = m0 + r;
                As[r * LDK + c] = (row < NN) ? (__bf16)x[(size_t)row * HH + c] : (__bf16)0.f;
            }
        } else {
            for (int i = threadIdx.x; i < 64 * 256; i += 128) {
                int r = i >> 8, c = i & 255;
                As[r * LDK + c] = vnb[(size_t)bsh[r] * HH + c];
            }
        }
        __syncthreads();
        #pragma unroll
        for (int ct = 0; ct < 16; ++ct)
            #pragma unroll
            for (int k0 = 0; k0 < 256; k0 += 32)
                acc[ct] = wmma_bf16(frag_lds(As, rowBase, k0),
                                    frag_gbl(Wa1T, ct * 16, kc * 256 + k0, HH2), acc[ct]);
    }
    // fused epilogue: +ba1, hardware tanh, dot with Wa2, cross-lane reduce
    float p[8];
    #pragma unroll
    for (int v = 0; v < 8; ++v) p[v] = 0.f;
    #pragma unroll
    for (int ct = 0; ct < 16; ++ct) {
        int col = ct * 16 + (lane & 15);
        float b = ba1[col], w2 = wa2[col];
        #pragma unroll
        for (int v = 0; v < 8; ++v)
            p[v] += hw_tanh(acc[ct][v] + b) * w2;
    }
    #pragma unroll
    for (int v = 0; v < 8; ++v) {
        p[v] += __shfl_xor(p[v], 1, 32);
        p[v] += __shfl_xor(p[v], 2, 32);
        p[v] += __shfl_xor(p[v], 4, 32);
        p[v] += __shfl_xor(p[v], 8, 32);
    }
    if ((lane & 15) == 0) {
        int rbase = m0 + rowBase + ((lane >> 4) << 3);
        #pragma unroll
        for (int v = 0; v < 8; ++v) {
            int row = rbase + v;
            if (row < NN) scores[row] = p[v];
        }
    }
}

// Online softmax reduction (max + sum of exp) --------------------------------
__device__ __forceinline__ void softmerge(float& m, float& s, float m2, float s2) {
    float M = fmaxf(m, m2);
    s = s * __expf(m - M) + s2 * __expf(m2 - M);
    m = M;
}
__global__ void k_red1(const float* __restrict__ sc, float* __restrict__ pm,
                       float* __restrict__ ps, int n) {
    __shared__ float sm[256], ss[256];
    float m = -3.0e38f, s = 0.f;
    for (int i = blockIdx.x * blockDim.x + threadIdx.x; i < n; i += gridDim.x * blockDim.x) {
        float v = sc[i];
        if (v > m) { s = s * __expf(m - v) + 1.f; m = v; }
        else         s += __expf(v - m);
    }
    sm[threadIdx.x] = m; ss[threadIdx.x] = s;
    __syncthreads();
    for (int off = 128; off > 0; off >>= 1) {
        if ((int)threadIdx.x < off) {
            float mm = sm[threadIdx.x], sv = ss[threadIdx.x];
            softmerge(mm, sv, sm[threadIdx.x + off], ss[threadIdx.x + off]);
            sm[threadIdx.x] = mm; ss[threadIdx.x] = sv;
        }
        __syncthreads();
    }
    if (threadIdx.x == 0) { pm[blockIdx.x] = sm[0]; ps[blockIdx.x] = ss[0]; }
}
__global__ void k_red2(const float* __restrict__ pm, const float* __restrict__ ps,
                       float* __restrict__ fin, int n) {
    __shared__ float sm[256], ss[256];
    float m = -3.0e38f, s = 0.f;
    for (int i = threadIdx.x; i < n; i += 256) softmerge(m, s, pm[i], ps[i]);
    sm[threadIdx.x] = m; ss[threadIdx.x] = s;
    __syncthreads();
    for (int off = 128; off > 0; off >>= 1) {
        if ((int)threadIdx.x < off) {
            float mm = sm[threadIdx.x], sv = ss[threadIdx.x];
            softmerge(mm, sv, sm[threadIdx.x + off], ss[threadIdx.x + off]);
            sm[threadIdx.x] = mm; ss[threadIdx.x] = sv;
        }
        __syncthreads();
    }
    if (threadIdx.x == 0) { fin[0] = sm[0]; fin[1] = 1.f / ss[0]; }
}
__global__ void k_weights(float* __restrict__ sc, const float* __restrict__ fin, int n) {
    int i = blockIdx.x * blockDim.x + threadIdx.x;
    if (i < n) sc[i] = __expf(sc[i] - fin[0]) * fin[1];
}

// vn_contribution[i] = vn_updated[batch[i]] * weight[i]   (float4 per thread)
__global__ void k_out(const float* __restrict__ vn, const int* __restrict__ batch,
                      const float* __restrict__ w, float* __restrict__ out) {
    size_t i = (size_t)blockIdx.x * blockDim.x + threadIdx.x;   // float4 index
    if (i >= (size_t)NN * (HH / 4)) return;
    int row = (int)(i >> 6);
    int c4  = (int)(i & 63);
    float4 v  = ((const float4*)(vn + (size_t)batch[row] * HH))[c4];
    float  ww = w[row];
    float4 o = { v.x * ww, v.y * ww, v.z * ww, v.w * ww };
    ((float4*)out)[i] = o;
}

// ---------------------------------------------------------------------------
extern "C" void kernel_launch(void* const* d_in, const int* in_sizes, int n_in,
                              void* d_out, int out_size, void* d_ws, size_t ws_size,
                              hipStream_t stream) {
    (void)in_sizes; (void)n_in; (void)out_size; (void)ws_size;
    const float* x      = (const float*)d_in[0];
    const int*   batch  = (const int*)  d_in[1];
    const float* ve     = (const float*)d_in[2];
    const float* W1     = (const float*)d_in[3];
    const float* b1     = (const float*)d_in[4];
    const float* gamma  = (const float*)d_in[5];
    const float* beta   = (const float*)d_in[6];
    const float* W2     = (const float*)d_in[7];
    const float* b2     = (const float*)d_in[8];
    const float* Wa1    = (const float*)d_in[9];
    const float* ba1    = (const float*)d_in[10];
    const float* Wa2    = (const float*)d_in[11];
    // d_in[12] = ba2: softmax-invariant, unused.

    float* out = (float*)d_out;
    float* vn_out = out + (size_t)NN * HH;   // second tuple element

    // workspace layout
    char* base = (char*)d_ws;
    size_t off = 0;
    auto alloc = [&](size_t bytes) { void* p = base + off; off = (off + bytes + 255) & ~(size_t)255; return p; };
    float*  sums   = (float*) alloc((size_t)GG * HH * 4 + GG * 4); // counts follow sums
    float*  counts = sums + (size_t)GG * HH;
    float*  a      = (float*) alloc((size_t)GG * HH * 4);
    float*  h1     = (float*) alloc((size_t)GG * HH2 * 4);
    float*  scale  = (float*) alloc(HH2 * 4);
    float*  shift  = (float*) alloc(HH2 * 4);
    __bf16* W1T    = (__bf16*)alloc((size_t)HH2 * HH * 2);
    __bf16* W2T    = (__bf16*)alloc((size_t)HH  * HH2 * 2);
    __bf16* Wa1T   = (__bf16*)alloc((size_t)HH  * HH2 * 2);
    __bf16* vnb    = (__bf16*)alloc((size_t)GG  * HH * 2);
    float*  scores = (float*) alloc((size_t)NN * 4);
    float*  pm     = (float*) alloc(512 * 4);
    float*  ps     = (float*) alloc(512 * 4);
    float*  fin    = (float*) alloc(2 * 4);

    const int nzero = GG * HH + GG;
    k_zero<<<(nzero + 255) / 256, 256, 0, stream>>>(sums, nzero);
    k_segsum<<<(NN + 127) / 128, 256, 0, stream>>>(x, batch, sums, counts);

    k_transpose_bf16<<<HH2, 256, 0, stream>>>(W1,  W1T,  HH,  HH2); // W1 [256,512]
    k_transpose_bf16<<<HH,  256, 0, stream>>>(W2,  W2T,  HH2, HH);  // W2 [512,256]
    k_transpose_bf16<<<HH,  256, 0, stream>>>(Wa1, Wa1T, HH2, HH);  // Wa1[512,256]

    k_vnpre<<<GG, HH, 0, stream>>>(sums, counts, ve, a);
    k_gemm1<<<dim3(GG / 64, 2), 128, 0, stream>>>(a, W1T, b1, h1);
    k_bnstats<<<HH2 / 128, 128, 0, stream>>>(h1, gamma, beta, scale, shift);
    k_gemm2<<<GG / 64, 128, 0, stream>>>(h1, scale, shift, W2T, b2, vn_out, vnb);

    k_scores<<<(NN + 63) / 64, 128, 0, stream>>>(x, batch, vnb, Wa1T, ba1, Wa2, scores);

    k_red1<<<512, 256, 0, stream>>>(scores, pm, ps, NN);
    k_red2<<<1, 256, 0, stream>>>(pm, ps, fin, 512);
    k_weights<<<(NN + 255) / 256, 256, 0, stream>>>(scores, fin, NN);
    k_out<<<(int)(((size_t)NN * (HH / 4) + 255) / 256), 256, 0, stream>>>(vn_out, batch, scores, out);
}
